// MultiHeadAttention_43516608643289
// MI455X (gfx1250) — matrix operand order repaired and checked by reading, not compiled, not run on hardware
//
#include <hip/hip_runtime.h>
#include <math.h>

typedef __attribute__((ext_vector_type(16))) __bf16          v16bf;
typedef __attribute__((ext_vector_type(8)))  __bf16          v8bf;
typedef __attribute__((ext_vector_type(8)))  float           v8f;
typedef __attribute__((ext_vector_type(8)))  unsigned short  v8u;
typedef __attribute__((ext_vector_type(4)))  unsigned int    u32x4;
typedef __attribute__((ext_vector_type(8)))  int             i32x8;
typedef __attribute__((ext_vector_type(4)))  int             i32x4;

#define D_IN   1024
#define D_OUT  1024
#define NHEAD  16
#define HDIM   64
#define SEQ    2048
#define BATCH  2
#define NTOK   (BATCH * SEQ)   // 4096
#define DQKV   (3 * D_OUT)     // 3072

__device__ __forceinline__ unsigned short f2bf(float f) {
  union { float f; unsigned u; } c; c.f = f;
  unsigned u = c.u;
  u += 0x7FFFu + ((u >> 16) & 1u);   // round-to-nearest-even
  return (unsigned short)(u >> 16);
}

// CDNA5 async global->LDS copy (16B per lane), tracked by ASYNCcnt.
__device__ __forceinline__ void async_g2l_b128(void* lds_ptr, const void* gptr) {
  unsigned lds_off = (unsigned)(size_t)lds_ptr;   // low 32 bits of generic = LDS offset
  asm volatile("global_load_async_to_lds_b128 %0, %1, off"
               :: "v"(lds_off), "v"(gptr) : "memory");
}

__device__ __forceinline__ void wait_async0() {
  asm volatile("s_wait_asynccnt 0x0" ::: "memory");
}

// Tensor Data Mover: DMA a 2D bf16 tile [tile_d1 rows x tile_d0 elems] from
// global (row stride `row_stride` elems) into LDS at `lds_off`, padding the
// LDS destination by 4 DWORDs after every 32 DWORDs (row 64 elems -> stride 72).
__device__ __forceinline__ void tdm_load_tile(unsigned lds_off, const void* gptr,
                                              unsigned tile_d0, unsigned tile_d1,
                                              unsigned row_stride) {
  unsigned long long ga = (unsigned long long)(size_t)gptr;
  u32x4 g0;
  g0[0] = 1u;                                            // count=1 (valid), user mode
  g0[1] = lds_off;                                       // lds_addr (bytes)
  g0[2] = (unsigned)(ga & 0xffffffffu);                  // global_addr[31:0]
  g0[3] = (unsigned)((ga >> 32) & 0x1ffffffu) | (2u << 30);  // addr[56:32] | type=2
  i32x8 g1;
  g1[0] = (int)((1u << 16)        // data_size = 2 bytes
              | (1u << 20)        // pad_enable
              | (4u << 22)        // pad_interval: 32 DWORDs
              | (3u << 25));      // pad_amount: 4 DWORDs
  g1[1] = (int)(tile_d0 << 16);   // tensor_dim0[15:0]
  g1[2] = (int)(tile_d1 << 16);   // tensor_dim0[31:16]=0 | tensor_dim1[15:0]
  g1[3] = (int)(tile_d0 << 16);   // tensor_dim1[31:16]=0 | tile_dim0
  g1[4] = (int)tile_d1;           // tile_dim1 | tile_dim2=0
  g1[5] = (int)row_stride;        // tensor_dim0_stride[31:0]
  g1[6] = 0;
  g1[7] = 0;
  i32x4 z4 = {0, 0, 0, 0};
#if __clang_major__ >= 23
  i32x8 z8 = {0, 0, 0, 0, 0, 0, 0, 0};
  __builtin_amdgcn_tensor_load_to_lds(g0, g1, z4, z4, z8, 0);
#else
  __builtin_amdgcn_tensor_load_to_lds(g0, g1, z4, z4, 0);
#endif
}

__device__ __forceinline__ void wait_tensor0() {
  __builtin_amdgcn_s_wait_tensorcnt(0);
}

// Build a 16-element bf16 fragment from two contiguous 8-element (16B) chunks.
__device__ __forceinline__ v16bf make_frag(const unsigned short* p_lo,
                                           const unsigned short* p_hi) {
  v8bf lo = *(const v8bf*)p_lo;
  v8bf hi = *(const v8bf*)p_hi;
  v16bf r;
#pragma unroll
  for (int i = 0; i < 8; ++i) { r[i] = lo[i]; r[i + 8] = hi[i]; }
  return r;
}

__global__ void cvt_f32_bf16(const float* __restrict__ src,
                             unsigned short* __restrict__ dst, int n) {
  int i = blockIdx.x * blockDim.x + threadIdx.x;
  if (i < n) dst[i] = f2bf(src[i]);
}

// C[M,N] = A[M,K] * B[N,K]^T   (A,B bf16; accumulate f32)
// Workgroup tile 128x128, wave tile 32x64 (8 WMMA / K-step).
// Double-buffered LDS staging via global_load_async_to_lds_b128.
__global__ __launch_bounds__(256)
void gemm_bf16_wmma(const unsigned short* __restrict__ A,
                    const unsigned short* __restrict__ B,
                    unsigned short* __restrict__ outBf,
                    float*          __restrict__ outF,
                    const float*    __restrict__ bias,
                    int M, int N, int K)
{
  __shared__ __attribute__((aligned(16))) unsigned short As[2][128][56];
  __shared__ __attribute__((aligned(16))) unsigned short Bs[2][128][56];

  const int mBase = blockIdx.y * 128;
  const int nBase = blockIdx.x * 128;
  const int t     = threadIdx.x;
  const int w     = t >> 5;
  const int lane  = t & 31;
  const int lo    = lane & 15;
  const int hi    = lane >> 4;
  const int waveM = w >> 1;   // 0..3 -> 32 rows each
  const int waveN = w & 1;    // 0..1 -> 64 cols each

  v8f acc[2][4];
#pragma unroll
  for (int a = 0; a < 2; ++a)
#pragma unroll
    for (int b = 0; b < 4; ++b)
#pragma unroll
      for (int e = 0; e < 8; ++e) acc[a][b][e] = 0.0f;

  // Fire async copies of one 32-wide K slab into buffer `buf`.
  auto stage = [&](int buf, int k0) {
#pragma unroll
    for (int it = 0; it < 2; ++it) {
      int c = t + 256 * it;
      int r = c >> 2, seg = c & 3;
      async_g2l_b128(&As[buf][r][seg * 8],
                     A + (size_t)(mBase + r) * K + k0 + seg * 8);
      async_g2l_b128(&Bs[buf][r][seg * 8],
                     B + (size_t)(nBase + r) * K + k0 + seg * 8);
    }
  };

  const int nk = K >> 5;
  stage(0, 0);
  wait_async0();
  __syncthreads();

  for (int i = 0; i < nk; ++i) {
    const int cur = i & 1;
    if (i + 1 < nk) {
      stage(1 - cur, (i + 1) * 32);               // overlap with WMMA below
      __builtin_prefetch(A + (size_t)(mBase + (t >> 1)) * K + (i + 1) * 32, 0, 3);
    }

    v16bf afrag[2], bfrag[4];
#pragma unroll
    for (int tm = 0; tm < 2; ++tm) {
      int r = waveM * 32 + tm * 16 + lo;
      afrag[tm] = make_frag(&As[cur][r][hi * 8], &As[cur][r][16 + hi * 8]);
    }
#pragma unroll
    for (int tn = 0; tn < 4; ++tn) {
      int c = waveN * 64 + tn * 16 + lo;
      bfrag[tn] = make_frag(&Bs[cur][c][hi * 8], &Bs[cur][c][16 + hi * 8]);
    }
#pragma unroll
    for (int tm = 0; tm < 2; ++tm)
#pragma unroll
      for (int tn = 0; tn < 4; ++tn)
        acc[tm][tn] = __builtin_amdgcn_wmma_f32_16x16x32_bf16(
            false, afrag[tm], false, bfrag[tn], (short)0, acc[tm][tn],
            false, false);

    wait_async0();       // my async writes into the other buffer are done
    __syncthreads();     // everyone: reads of cur done, writes of next done
  }

#pragma unroll
  for (int tm = 0; tm < 2; ++tm) {
#pragma unroll
    for (int tn = 0; tn < 4; ++tn) {
      int n  = nBase + waveN * 64 + tn * 16 + lo;
      int m0 = mBase + waveM * 32 + tm * 16 + hi * 8;
      float bv = bias ? bias[n] : 0.0f;
#pragma unroll
      for (int i = 0; i < 8; ++i) {
        float v = acc[tm][tn][i] + bv;
        size_t idx = (size_t)(m0 + i) * N + n;
        if (outF) outF[idx] = v;
        else      outBf[idx] = f2bf(v);
      }
    }
  }
}

// Fused causal flash-attention. One block = (batch b, head h, 128-query slab);
// 8 waves, each owns a 16-query tile; K staged by the Tensor Data Mover,
// V staged transposed by per-lane loads; all shared through LDS.
__global__ __launch_bounds__(256)
void attn_wmma(const unsigned short* __restrict__ qkv,   // [NTOK][3072]
               unsigned short* __restrict__ ctx)         // [NTOK][1024]
{
  __shared__ __attribute__((aligned(16))) unsigned short Ks[32][72];     // [key][dim] (TDM-padded)
  __shared__ __attribute__((aligned(16))) unsigned short Vt[64][40];     // [dim][key]
  __shared__ __attribute__((aligned(16))) unsigned short Ps[8][16][40];  // per-wave P

  const int blk  = blockIdx.x;
  const int qblk = blk & 15;          // SEQ/128 = 16
  const int h    = (blk >> 4) & 15;
  const int b    = blk >> 8;

  const int t    = threadIdx.x;
  const int w    = t >> 5;
  const int lane = t & 31;
  const int lo   = lane & 15;
  const int hi   = lane >> 4;
  const int qrow0 = qblk * 128 + w * 16;

  // Q tile 16x64 -> two 16x32 A-fragments, loaded once from global.
  v16bf qf[2];
  {
    const size_t tok = (size_t)(b * SEQ + qrow0 + lo);
    const unsigned short* qp = qkv + tok * DQKV + h * HDIM;
#pragma unroll
    for (int ks = 0; ks < 2; ++ks)
      qf[ks] = make_frag(qp + ks * 32 + hi * 8, qp + ks * 32 + 16 + hi * 8);
  }

  float mrun[8], lsum[8];
  v8f oacc[4];
#pragma unroll
  for (int i = 0; i < 8; ++i) { mrun[i] = -1e30f; lsum[i] = 0.0f; }
#pragma unroll
  for (int d = 0; d < 4; ++d)
#pragma unroll
    for (int i = 0; i < 8; ++i) oacc[d][i] = 0.0f;

  const int nsteps = qblk * 4 + 4;    // 32-key steps needed by deepest wave
  for (int s = 0; s < nsteps; ++s) {
    const int kbase = s * 32;
    // K tile (32 keys x 64 dims) via TDM, issued once by wave 0.
    if (w == 0) {
      tdm_load_tile((unsigned)(size_t)(void*)&Ks[0][0],
                    qkv + (size_t)(b * SEQ + kbase) * DQKV + D_OUT + h * HDIM,
                    HDIM, 32, DQKV);
    }
    {   // V transposed (32 keys x 64 dims), per-lane
      int r = t >> 3, seg = t & 7;
      const size_t tok = (size_t)(b * SEQ + kbase + r);
      v8u vv = *(const v8u*)(qkv + tok * DQKV + 2 * D_OUT + h * HDIM + seg * 8);
#pragma unroll
      for (int j = 0; j < 8; ++j) Vt[seg * 8 + j][r] = vv[j];
    }
    if (w == 0) wait_tensor0();
    __syncthreads();

    if (kbase <= qrow0 + 15) {        // wave-uniform causal cutoff
      // S = Q*K^T : two 16-key tiles, K-dim 64 = 2 WMMA steps each
      v8f sc[2];
#pragma unroll
      for (int kt = 0; kt < 2; ++kt) {
#pragma unroll
        for (int e = 0; e < 8; ++e) sc[kt][e] = 0.0f;
#pragma unroll
        for (int ks = 0; ks < 2; ++ks) {
          v16bf bf = make_frag(&Ks[kt * 16 + lo][ks * 32 + hi * 8],
                               &Ks[kt * 16 + lo][ks * 32 + 16 + hi * 8]);
          sc[kt] = __builtin_amdgcn_wmma_f32_16x16x32_bf16(
              false, qf[ks], false, bf, (short)0, sc[kt], false, false);
        }
      }
      // Causal mask + online softmax (row m = i + 8*hi; col n = lo).
      float pm[2][8];
#pragma unroll
      for (int i = 0; i < 8; ++i) {
        const int qabs = qrow0 + i + 8 * hi;
#pragma unroll
        for (int kt = 0; kt < 2; ++kt) {
          const int kabs = kbase + kt * 16 + lo;
          float sv = sc[kt][i];
          if (kabs > qabs) sv = -1e30f;
          sc[kt][i] = sv;
        }
        float mc = fmaxf(sc[0][i], sc[1][i]);
#pragma unroll
        for (int off = 1; off < 16; off <<= 1)
          mc = fmaxf(mc, __shfl_xor(mc, off, 32));
        const float mnew = fmaxf(mrun[i], mc);
        const float corr = __expf(mrun[i] - mnew);
        mrun[i] = mnew;
        float p0 = __expf(sc[0][i] - mnew);
        float p1 = __expf(sc[1][i] - mnew);
        float ps = p0 + p1;
#pragma unroll
        for (int off = 1; off < 16; off <<= 1)
          ps += __shfl_xor(ps, off, 32);
        lsum[i] = lsum[i] * corr + ps;
#pragma unroll
        for (int d = 0; d < 4; ++d) oacc[d][i] *= corr;
        pm[0][i] = p0; pm[1][i] = p1;
      }
      // Reshape P (C-layout) -> A-fragment via per-wave LDS round trip.
#pragma unroll
      for (int i = 0; i < 8; ++i)
#pragma unroll
        for (int kt = 0; kt < 2; ++kt)
          Ps[w][i + 8 * hi][kt * 16 + lo] = f2bf(pm[kt][i]);

      v16bf pf = make_frag(&Ps[w][lo][hi * 8], &Ps[w][lo][16 + hi * 8]);
#pragma unroll
      for (int d = 0; d < 4; ++d) {
        v16bf vf = make_frag(&Vt[d * 16 + lo][hi * 8],
                             &Vt[d * 16 + lo][16 + hi * 8]);
        oacc[d] = __builtin_amdgcn_wmma_f32_16x16x32_bf16(
            false, pf, false, vf, (short)0, oacc[d], false, false);
      }
    }
    __syncthreads();
  }

#pragma unroll
  for (int i = 0; i < 8; ++i) {
    const float rinv = 1.0f / lsum[i];
    const size_t tok = (size_t)(b * SEQ + qrow0 + i + 8 * hi);
#pragma unroll
    for (int d = 0; d < 4; ++d)
      ctx[tok * D_OUT + h * HDIM + d * 16 + lo] = f2bf(oacc[d][i] * rinv);
  }
}

extern "C" void kernel_launch(void* const* d_in, const int* in_sizes, int n_in,
                              void* d_out, int out_size, void* d_ws, size_t ws_size,
                              hipStream_t stream) {
  const float* x    = (const float*)d_in[0];   // [2,2048,1024]
  const float* Wqkv = (const float*)d_in[1];   // [3072,1024]
  const float* Wout = (const float*)d_in[2];   // [1024,1024]
  const float* bout = (const float*)d_in[3];   // [1024]
  float* out = (float*)d_out;                  // [2,2048,1024]

  unsigned short* xb   = (unsigned short*)d_ws;
  unsigned short* wqb  = xb   + (size_t)NTOK * D_IN;
  unsigned short* wob  = wqb  + (size_t)DQKV * D_IN;
  unsigned short* qkvb = wob  + (size_t)D_OUT * D_OUT;
  unsigned short* ctxb = qkvb + (size_t)NTOK * DQKV;

  const int nx = NTOK * D_IN;
  const int nq = DQKV * D_IN;
  const int nw = D_OUT * D_OUT;
  cvt_f32_bf16<<<nx / 256, 256, 0, stream>>>(x,    xb,  nx);
  cvt_f32_bf16<<<nq / 256, 256, 0, stream>>>(Wqkv, wqb, nq);
  cvt_f32_bf16<<<nw / 256, 256, 0, stream>>>(Wout, wob, nw);

  // qkv = x @ W_qkv^T   (bf16 out to workspace)
  gemm_bf16_wmma<<<dim3(DQKV / 128, NTOK / 128), 256, 0, stream>>>(
      xb, wqb, qkvb, nullptr, nullptr, NTOK, DQKV, D_IN);

  // fused causal attention -> ctx (bf16)
  attn_wmma<<<dim3(BATCH * NHEAD * (SEQ / 128)), 256, 0, stream>>>(qkvb, ctxb);

  // out = ctx @ W_out^T + b_out  (f32 to d_out)
  gemm_bf16_wmma<<<dim3(D_OUT / 128, NTOK / 128), 256, 0, stream>>>(
      ctxb, wob, nullptr, out, bout, NTOK, D_OUT, D_OUT);
}
